// RoPEMultiheadAttention_28132035789442
// MI455X (gfx1250) — matrix operand order repaired and checked by reading, not compiled, not run on hardware
//
#include <hip/hip_runtime.h>
#include <hip/hip_bf16.h>
#include <math.h>

#define D_MODEL  1024
#define NHEADS   16
#define HEAD_DIM 64
#define BATCH    4
#define SEQ      2048
#define BN       (BATCH * SEQ)   // 8192

typedef __bf16 bf16;
typedef __attribute__((ext_vector_type(16))) bf16  v16bf;
typedef __attribute__((ext_vector_type(8)))  bf16  v8bf;
typedef __attribute__((ext_vector_type(4)))  bf16  v4bf;
typedef __attribute__((ext_vector_type(8)))  float v8f;

#define WMMA_BF16(a, b, c) \
    __builtin_amdgcn_wmma_f32_16x16x32_bf16(false, (a), false, (b), (short)0, (c), false, false)

static __device__ __forceinline__ v16bf make16(v8bf lo, v8bf hi) {
    return __builtin_shufflevector(lo, hi, 0,1,2,3,4,5,6,7,8,9,10,11,12,13,14,15);
}

// A operand 16x32 bf16: lane holds row (l&15); K chunks at half*8 and 16+half*8.
static __device__ __forceinline__ v16bf load_a(const bf16* row, int half) {
    v8bf lo = *reinterpret_cast<const v8bf*>(row + half * 8);
    v8bf hi = *reinterpret_cast<const v8bf*>(row + 16 + half * 8);
    return make16(lo, hi);
}

// B operand 32x16 bf16: lane holds column (l&15); K chunks at half*8 and 16+half*8.
static __device__ __forceinline__ v16bf load_b(const bf16* col, int half) {
    v8bf lo = *reinterpret_cast<const v8bf*>(col + half * 8);
    v8bf hi = *reinterpret_cast<const v8bf*>(col + 16 + half * 8);
    return make16(lo, hi);
}

// ---------------------------------------------------------------- fp32 -> bf16
__global__ void cvt_f32_bf16(const float* __restrict__ src, bf16* __restrict__ dst, int n4) {
    int i = blockIdx.x * blockDim.x + threadIdx.x;
    if (i < n4) {
        float4 f = reinterpret_cast<const float4*>(src)[i];
        v4bf o;
        o.x = (bf16)f.x; o.y = (bf16)f.y; o.z = (bf16)f.z; o.w = (bf16)f.w;
        reinterpret_cast<v4bf*>(dst)[i] = o;
    }
}

// --------------------------------------------- QKV projection + bias + RoPE
// grid: (BN/32, NHEADS, 3)  block: 32. One wave computes a 32x64 head tile
// (two 16-row A tiles share each B operand -> 8 WMMA per k-step).
__global__ __launch_bounds__(32) void qkv_rope_kernel(
    const bf16* __restrict__ xb,
    const bf16* __restrict__ wq, const bf16* __restrict__ wk, const bf16* __restrict__ wv,
    const float* __restrict__ bq, const float* __restrict__ bk, const float* __restrict__ bv,
    bf16* __restrict__ qout, bf16* __restrict__ kout, bf16* __restrict__ vTout)
{
    const int lane = threadIdx.x & 31;
    const int l15  = lane & 15;
    const int half = lane >> 4;
    const int mbase = blockIdx.x * 32;
    const int h     = blockIdx.y;
    const int proj  = blockIdx.z;
    const bf16*  W    = (proj == 0) ? wq : (proj == 1) ? wk : wv;
    const float* bias = (proj == 0) ? bq : (proj == 1) ? bk : bv;
    const int nbase = h * HEAD_DIM;

    v8f zero = {0.f,0.f,0.f,0.f,0.f,0.f,0.f,0.f};
    v8f acc[2][4] = {{zero, zero, zero, zero}, {zero, zero, zero, zero}};

    const bf16* arow0 = xb + (size_t)(mbase + l15) * D_MODEL;
    const bf16* arow1 = xb + (size_t)(mbase + 16 + l15) * D_MODEL;
    for (int kk = 0; kk < D_MODEL; kk += 32) {
        v16bf a0 = load_a(arow0 + kk, half);
        v16bf a1 = load_a(arow1 + kk, half);
#pragma unroll
        for (int t = 0; t < 4; ++t) {
            const bf16* col = W + (size_t)(nbase + t * 16 + l15) * D_MODEL + kk;
            v16bf b = load_b(col, half);
            acc[0][t] = WMMA_BF16(a0, b, acc[0][t]);
            acc[1][t] = WMMA_BF16(a1, b, acc[1][t]);
        }
    }

#pragma unroll
    for (int qg = 0; qg < 2; ++qg) {
        float res[4][8];
#pragma unroll
        for (int t = 0; t < 4; ++t) {
            float bs = bias[nbase + t * 16 + l15];
#pragma unroll
            for (int j = 0; j < 8; ++j) res[t][j] = acc[qg][t][j] + bs;
        }

        if (proj < 2) {  // RoPE on Q,K: pairs (d, d+32) = accumulators (t, t+2)
#pragma unroll
            for (int t = 0; t < 2; ++t) {
                const int jj = t * 16 + l15;                           // freq idx 0..31
                const float freq = __expf(-(float)jj * 0.3597789165f); // ln(1e5)/32
#pragma unroll
                for (int j = 0; j < 8; ++j) {
                    int g = mbase + qg * 16 + half * 8 + j;
                    float pos = (float)(g & (SEQ - 1));
                    float ang = pos * freq;
                    float c = __cosf(ang), s = __sinf(ang);
                    float x1 = res[t][j], x2 = res[t + 2][j];
                    res[t][j]     = x1 * c - x2 * s;
                    res[t + 2][j] = x1 * s + x2 * c;
                }
            }
        }

        if (proj == 2) {  // V stored transposed: [b, h, d, pos]
#pragma unroll
            for (int t = 0; t < 4; ++t) {
                int d = t * 16 + l15;
#pragma unroll
                for (int j = 0; j < 8; ++j) {
                    int g = mbase + qg * 16 + half * 8 + j;
                    int b = g >> 11, pos = g & (SEQ - 1);
                    vTout[((size_t)(b * NHEADS + h) * HEAD_DIM + d) * SEQ + pos] = (bf16)res[t][j];
                }
            }
        } else {          // Q,K stored [b, h, pos, d]
            bf16* out = (proj == 0) ? qout : kout;
#pragma unroll
            for (int t = 0; t < 4; ++t) {
                int d = t * 16 + l15;
#pragma unroll
                for (int j = 0; j < 8; ++j) {
                    int g = mbase + qg * 16 + half * 8 + j;
                    int b = g >> 11, pos = g & (SEQ - 1);
                    out[((size_t)(b * NHEADS + h) * SEQ + pos) * HEAD_DIM + d] = (bf16)res[t][j];
                }
            }
        }
    }
}

// ----------------------------------------------------- flash attention core
// grid: (SEQ/32, BATCH*NHEADS)  block: 32. One wave owns 32 queries; K/V
// operands are loaded once per 32-key tile and reused by both query tiles.
__global__ __launch_bounds__(32) void attn_kernel(
    const bf16* __restrict__ q, const bf16* __restrict__ k,
    const bf16* __restrict__ vT, bf16* __restrict__ ao)
{
    const int lane = threadIdx.x & 31;
    const int l15  = lane & 15;
    const int half = lane >> 4;
    const int qt = blockIdx.x;
    const int bh = blockIdx.y;
    const int b = bh >> 4, h = bh & 15;
    const float scale = 0.125f;  // 1/sqrt(64)

    __shared__ bf16 pbuf[32 * 32];

    v16bf aq[2][2];
#pragma unroll
    for (int qg = 0; qg < 2; ++qg) {
        const bf16* qrow = q + ((size_t)bh * SEQ + qt * 32 + qg * 16 + l15) * HEAD_DIM;
        aq[qg][0] = load_a(qrow, half);       // d 0..31
        aq[qg][1] = load_a(qrow + 32, half);  // d 32..63
    }

    v8f zero = {0.f,0.f,0.f,0.f,0.f,0.f,0.f,0.f};
    v8f acc[2][4] = {{zero, zero, zero, zero}, {zero, zero, zero, zero}};
    float mrow[2][8], lrow[2][8];
#pragma unroll
    for (int qg = 0; qg < 2; ++qg)
#pragma unroll
        for (int j = 0; j < 8; ++j) { mrow[qg][j] = -INFINITY; lrow[qg][j] = 0.f; }

    for (int kt = 0; kt < SEQ; kt += 32) {
        v16bf bk[2][2];
#pragma unroll
        for (int s = 0; s < 2; ++s) {
            const bf16* kcol = k + ((size_t)bh * SEQ + kt + s * 16 + l15) * HEAD_DIM;
            bk[s][0] = load_b(kcol, half);
            bk[s][1] = load_b(kcol + 32, half);
        }
        float sc[2][2][8];
#pragma unroll
        for (int qg = 0; qg < 2; ++qg)
#pragma unroll
            for (int s = 0; s < 2; ++s) {
                v8f sa = zero;
                sa = WMMA_BF16(aq[qg][0], bk[s][0], sa);
                sa = WMMA_BF16(aq[qg][1], bk[s][1], sa);
#pragma unroll
                for (int j = 0; j < 8; ++j) sc[qg][s][j] = sa[j] * scale;
            }

        // online softmax: row stats live across the 16 lanes of each half-wave
        float alpha[2][8];
#pragma unroll
        for (int qg = 0; qg < 2; ++qg)
#pragma unroll
            for (int j = 0; j < 8; ++j) {
                float r = fmaxf(sc[qg][0][j], sc[qg][1][j]);
#pragma unroll
                for (int off = 1; off < 16; off <<= 1) r = fmaxf(r, __shfl_xor(r, off, 32));
                float nm = fmaxf(mrow[qg][j], r);
                alpha[qg][j] = __expf(mrow[qg][j] - nm);
                mrow[qg][j] = nm;
                sc[qg][0][j] = __expf(sc[qg][0][j] - nm);
                sc[qg][1][j] = __expf(sc[qg][1][j] - nm);
                float su = sc[qg][0][j] + sc[qg][1][j];
#pragma unroll
                for (int off = 1; off < 16; off <<= 1) su += __shfl_xor(su, off, 32);
                lrow[qg][j] = lrow[qg][j] * alpha[qg][j] + su;
            }
#pragma unroll
        for (int qg = 0; qg < 2; ++qg)
#pragma unroll
            for (int t = 0; t < 4; ++t)
#pragma unroll
                for (int j = 0; j < 8; ++j) acc[qg][t][j] *= alpha[qg][j];

        // P: C-layout (lane=key col) -> A-layout (lane=query row) via LDS
#pragma unroll
        for (int qg = 0; qg < 2; ++qg)
#pragma unroll
            for (int s = 0; s < 2; ++s)
#pragma unroll
                for (int j = 0; j < 8; ++j)
                    pbuf[(qg * 16 + half * 8 + j) * 32 + s * 16 + l15] = (bf16)sc[qg][s][j];
        __syncthreads();
        v16bf pa[2];
#pragma unroll
        for (int qg = 0; qg < 2; ++qg)
            pa[qg] = load_a(pbuf + (qg * 16 + l15) * 32, half);
        __syncthreads();

#pragma unroll
        for (int t = 0; t < 4; ++t) {
            const bf16* vcol = vT + ((size_t)bh * HEAD_DIM + t * 16 + l15) * SEQ + kt;
            v16bf bv = load_b(vcol, half);
            acc[0][t] = WMMA_BF16(pa[0], bv, acc[0][t]);
            acc[1][t] = WMMA_BF16(pa[1], bv, acc[1][t]);
        }
    }

#pragma unroll
    for (int qg = 0; qg < 2; ++qg) {
        float inv[8];
#pragma unroll
        for (int j = 0; j < 8; ++j) inv[j] = 1.f / lrow[qg][j];
#pragma unroll
        for (int t = 0; t < 4; ++t) {
            int d = t * 16 + l15;
#pragma unroll
            for (int j = 0; j < 8; ++j) {
                int pos = qt * 32 + qg * 16 + half * 8 + j;
                ao[(size_t)(b * SEQ + pos) * D_MODEL + h * HEAD_DIM + d] =
                    (bf16)(acc[qg][t][j] * inv[j]);
            }
        }
    }
}

// ------------------------------------------------------- output projection
// grid: (BN/32, D_MODEL/64)  block: 32. 32x64 tile per wave.
__global__ __launch_bounds__(32) void oproj_kernel(
    const bf16* __restrict__ ab, const bf16* __restrict__ wo,
    const float* __restrict__ bo, float* __restrict__ out)
{
    const int lane = threadIdx.x & 31;
    const int l15  = lane & 15;
    const int half = lane >> 4;
    const int mbase = blockIdx.x * 32;
    const int nbase = blockIdx.y * 64;

    v8f zero = {0.f,0.f,0.f,0.f,0.f,0.f,0.f,0.f};
    v8f acc[2][4] = {{zero, zero, zero, zero}, {zero, zero, zero, zero}};

    const bf16* arow0 = ab + (size_t)(mbase + l15) * D_MODEL;
    const bf16* arow1 = ab + (size_t)(mbase + 16 + l15) * D_MODEL;
    for (int kk = 0; kk < D_MODEL; kk += 32) {
        v16bf a0 = load_a(arow0 + kk, half);
        v16bf a1 = load_a(arow1 + kk, half);
#pragma unroll
        for (int t = 0; t < 4; ++t) {
            const bf16* col = wo + (size_t)(nbase + t * 16 + l15) * D_MODEL + kk;
            v16bf b = load_b(col, half);
            acc[0][t] = WMMA_BF16(a0, b, acc[0][t]);
            acc[1][t] = WMMA_BF16(a1, b, acc[1][t]);
        }
    }
#pragma unroll
    for (int qg = 0; qg < 2; ++qg)
#pragma unroll
        for (int t = 0; t < 4; ++t) {
            int n = nbase + t * 16 + l15;
            float bs = bo[n];
#pragma unroll
            for (int j = 0; j < 8; ++j) {
                int g = mbase + qg * 16 + half * 8 + j;
                out[(size_t)g * D_MODEL + n] = acc[qg][t][j] + bs;
            }
        }
}

extern "C" void kernel_launch(void* const* d_in, const int* in_sizes, int n_in,
                              void* d_out, int out_size, void* d_ws, size_t ws_size,
                              hipStream_t stream) {
    const float* x  = (const float*)d_in[0];
    const float* Wq = (const float*)d_in[1];
    const float* bq = (const float*)d_in[2];
    const float* Wk = (const float*)d_in[3];
    const float* bk = (const float*)d_in[4];
    const float* Wv = (const float*)d_in[5];
    const float* bv = (const float*)d_in[6];
    const float* Wo = (const float*)d_in[7];
    const float* bo = (const float*)d_in[8];
    float* out = (float*)d_out;

    const size_t nx = (size_t)BN * D_MODEL;       // 8388608
    const size_t nw = (size_t)D_MODEL * D_MODEL;  // 1048576
    char* p = (char*)d_ws;
    bf16* xb  = (bf16*)p; p += nx * 2;
    bf16* wqb = (bf16*)p; p += nw * 2;
    bf16* wkb = (bf16*)p; p += nw * 2;
    bf16* wvb = (bf16*)p; p += nw * 2;
    bf16* wob = (bf16*)p; p += nw * 2;
    bf16* qr  = (bf16*)p; p += nx * 2;  // [B,H,N,Hd] after RoPE
    bf16* kr  = (bf16*)p; p += nx * 2;  // [B,H,N,Hd] after RoPE
    bf16* vT  = (bf16*)p; p += nx * 2;  // [B,H,Hd,N]
    bf16* ao  = (bf16*)p; p += nx * 2;  // [B,N,D]

    cvt_f32_bf16<<<(unsigned)(nx / 1024), 256, 0, stream>>>(x,  xb,  (int)(nx / 4));
    cvt_f32_bf16<<<(unsigned)(nw / 1024), 256, 0, stream>>>(Wq, wqb, (int)(nw / 4));
    cvt_f32_bf16<<<(unsigned)(nw / 1024), 256, 0, stream>>>(Wk, wkb, (int)(nw / 4));
    cvt_f32_bf16<<<(unsigned)(nw / 1024), 256, 0, stream>>>(Wv, wvb, (int)(nw / 4));
    cvt_f32_bf16<<<(unsigned)(nw / 1024), 256, 0, stream>>>(Wo, wob, (int)(nw / 4));

    qkv_rope_kernel<<<dim3(BN / 32, NHEADS, 3), 32, 0, stream>>>(
        xb, wqb, wkb, wvb, bq, bk, bv, qr, kr, vT);

    attn_kernel<<<dim3(SEQ / 32, BATCH * NHEADS), 32, 0, stream>>>(qr, kr, vT, ao);

    oproj_kernel<<<dim3(BN / 32, D_MODEL / 64), 32, 0, stream>>>(ao, wob, bo, out);
}
